// Network_45208825757863
// MI455X (gfx1250) — matrix operand order repaired and checked by reading, not compiled, not run on hardware
//
#include <hip/hip_runtime.h>
#include <hip/hip_bf16.h>
#include <stdint.h>

// ---------------------------------------------------------------------------
// Problem constants (from the reference)
// ---------------------------------------------------------------------------
#define N_RAYS   1024
#define S_SAMP   128          // 64 coarse + 64 importance, merged+sorted
#define NPTS     (N_RAYS * S_SAMP)   // 131072
#define HID      256
#define KPAD     288          // padded K for the 265/259-wide MLP inputs
#define PTS_PER_WG 32         // 32 points * 4 rows (h,tx,ty,tz) = 128 M-rows
#define MROWS    128

typedef unsigned short u16;
typedef __attribute__((ext_vector_type(16))) __bf16 v16bf;
typedef __attribute__((ext_vector_type(8)))  float  v8f;
typedef __attribute__((ext_vector_type(4)))  unsigned int v4u;
typedef __attribute__((ext_vector_type(8)))  int  v8i;
typedef __attribute__((ext_vector_type(4)))  int  v4i;

union Frag32 { uint4 q[2]; v16bf v; };

#if __has_builtin(__builtin_amdgcn_tensor_load_to_lds) && \
    __has_builtin(__builtin_amdgcn_s_wait_tensorcnt)
#define USE_TDM 1
#else
#define USE_TDM 0
#endif

// ---------------------------------------------------------------------------
// Workspace layout (bytes)
// ---------------------------------------------------------------------------
#define O_DALL    0u                               // 128 f32
#define O_DIR     512u                             // 1024*3 f32
#define O_PTS     (O_DIR + 12288u)                 // NPTS*3 f32
#define O_PP      (O_PTS + 1572864u)               // NPTS*8 f32 (sdf,rgb3,sem3,pad)
#define O_SWB     (O_PP  + 4194304u)               // 7*256*256 bf16
#define O_FEATW   (O_SWB + 917504u)                // 256*256 bf16
#define O_SDFCOL  (O_FEATW + 131072u)              // 256 f32
#define O_RADINW  (O_SDFCOL + 1024u)               // 288*256 bf16
#define O_RWB     (O_RADINW + 147456u)             // 2*256*256 bf16
#define O_SEMINW  (O_RWB + 262144u)                // 288*256 bf16

// Dynamic LDS: Wl[288*256] + b0[128*288] + b1[128*288] (bf16) + small f32 area
#define SMEM_BYTES (288*256*2 + 2*(MROWS*KPAD*2) + 2048)

// ---------------------------------------------------------------------------
// Helpers
// ---------------------------------------------------------------------------
__device__ __forceinline__ u16 f2bf(float f) {
    unsigned int u = __float_as_uint(f);
    unsigned int r = 0x7fffu + ((u >> 16) & 1u);
    return (u16)((u + r) >> 16);
}
__device__ __forceinline__ float bf2f(u16 h) {
    return __uint_as_float(((unsigned int)h) << 16);
}
__device__ __forceinline__ float softplusf(float x) {
    return fmaxf(x, 0.f) + log1pf(__expf(-fabsf(x)));
}

// ---------------------------------------------------------------------------
// Stage one weight matrix [rows][256] bf16 from global into LDS at offset 0.
// Preferred path: Tensor Data Mover (one descriptor, issued by wave 0,
// completion via TENSORcnt). Fallback: VGPR copy loop.
// Caller must __syncthreads() afterwards (and have synced before so no wave
// still reads the previous layer's weights).
// ---------------------------------------------------------------------------
__device__ __forceinline__ void stage_w(u16* __restrict__ dstLds,
                                        const u16* __restrict__ src,
                                        int elems, int tid, int wave) {
#if USE_TDM
    (void)dstLds; (void)tid;
    if (wave == 0) {
        const unsigned int rows = (unsigned int)(elems >> 8);   // 256 cols/row
        unsigned long long ga = (unsigned long long)(uintptr_t)src;
        // D# group 0: count=1 | lds_addr=0 | global_addr | type=2
        v4u g0 = { 1u,
                   0u,
                   (unsigned int)ga,
                   ((unsigned int)((ga >> 32) & 0x01ffffffull)) | 0x80000000u };
        // D# group 1: data_size=2B, tensor 256 x rows, tile 256 x rows,
        //             dim0 stride = 256 elements
        v8i g1 = { (int)0x00010000u,                 // data_size=1 (2 bytes)
                   (int)(256u << 16),                // tensor_dim0 = 256 (lo16 @bit48)
                   (int)((rows & 0xffffu) << 16),    // tensor_dim1 lo16 @bit80
                   (int)(256u << 16),                // tile_dim0 = 256 @bit112
                   (int)rows,                        // tile_dim1 @bit128
                   256,                              // tensor_dim0_stride lo32
                   0, 0 };
        v4i z4 = { 0, 0, 0, 0 };
        v8i z8 = { 0, 0, 0, 0, 0, 0, 0, 0 };
        __builtin_amdgcn_tensor_load_to_lds(g0, g1, z4, z4, z8, 0);
        __builtin_amdgcn_s_wait_tensorcnt(0);
    }
#else
    int n = elems >> 3;                       // uint4 = 8 bf16
    const uint4* s = (const uint4*)src;
    uint4* d = (uint4*)dstLds;
    for (int i = tid; i < n; i += 256) d[i] = s[i];
#endif
}

// ---------------------------------------------------------------------------
// WMMA GEMM over LDS tiles.
//   A  : [Mtiles*16][KPAD] bf16 in LDS (row-major, stride KPAD)
//   Wl : [KSTEPS*32][256]  bf16 in LDS (row-major)
//   Out: bf16 in LDS, stride KPAD, written at (outRowOff+row, outColOff+col)
// MODE 0: SDF layer  -> row%4==0: softplus(z+b); rows 1..3: tangent * (1-e^-h)
// MODE 1: ReLU(z+b)  -> bf16
// MODE 2: z+b        -> bf16
// ---------------------------------------------------------------------------
template<int MODE, int KSTEPS>
__device__ __forceinline__ void wmma_gemm(const u16* __restrict__ A,
                                          const u16* __restrict__ Wl,
                                          u16* __restrict__ Out,
                                          int outColOff, int outRowOff,
                                          const float* __restrict__ bias,
                                          int Mtiles, int wave, int lane) {
    const int LDA = KPAD;
    const int tiles = Mtiles * 16;
    for (int t = wave; t < tiles; t += 8) {
        const int mt = t >> 4, nt = t & 15;
        v8f c = {};
        const int arow = mt * 16 + (lane & 15);
        const int koff = (lane < 16) ? 0 : 8;      // A-layout K interleave
        const u16* abase = A + (size_t)arow * LDA + koff;
        const u16* bbase = Wl + (size_t)koff * 256 + nt * 16 + (lane & 15);  // B: lane = N col, K interleave as A
#pragma unroll
        for (int ks = 0; ks < KSTEPS; ++ks) {
            Frag32 af, bf_;
            const u16* ap = abase + ks * 32;
            af.q[0] = *(const uint4*)(ap);          // K {0..7}/{8..15}
            af.q[1] = *(const uint4*)(ap + 16);     // K {16..23}/{24..31}
            const u16* bp = bbase + (size_t)ks * 32 * 256;
            unsigned int bw[8];
#pragma unroll
            for (int j = 0; j < 8; ++j) {
                const int kk = (j < 4) ? 2 * j : 16 + 2 * (j - 4);
                bw[j] = (unsigned int)bp[(size_t)kk * 256] | ((unsigned int)bp[(size_t)(kk + 1) * 256] << 16);
            }
            bf_.q[0] = make_uint4(bw[0], bw[1], bw[2], bw[3]);
            bf_.q[1] = make_uint4(bw[4], bw[5], bw[6], bw[7]);
            c = __builtin_amdgcn_wmma_f32_16x16x32_bf16(
                    false, af.v, false, bf_.v, (short)0, c, false, false);
        }
        const int col = nt * 16 + (lane & 15);
        const float bc = bias[col];
        const int rbase = outRowOff + mt * 16 + ((lane < 16) ? 0 : 8);
        u16* ob = Out + (size_t)rbase * LDA + outColOff + col;
        float cc[8];
#pragma unroll
        for (int r0 = 0; r0 < 8; ++r0) cc[r0] = c[r0];
        if (MODE == 0) {
            // rows grouped per point: [h, tx, ty, tz] all in this lane
#pragma unroll
            for (int g = 0; g < 8; g += 4) {
                float z = cc[g] + bc;
                float h = softplusf(z);
                float m = 1.f - __expf(-h);        // softplus' = sigmoid(z)
                ob[(size_t)(g + 0) * LDA] = f2bf(h);
                ob[(size_t)(g + 1) * LDA] = f2bf(cc[g + 1] * m);
                ob[(size_t)(g + 2) * LDA] = f2bf(cc[g + 2] * m);
                ob[(size_t)(g + 3) * LDA] = f2bf(cc[g + 3] * m);
            }
        } else {
#pragma unroll
            for (int r0 = 0; r0 < 8; ++r0) {
                float v = cc[r0] + bc;
                if (MODE == 1) v = fmaxf(v, 0.f);
                ob[(size_t)r0 * LDA] = f2bf(v);
            }
        }
    }
}

// ---------------------------------------------------------------------------
// Kernel: merged sorted depth vector d_all[128] (constant for every ray)
// ---------------------------------------------------------------------------
__global__ void k_dall(float* __restrict__ dall) {
    int i = threadIdx.x;
    if (i >= 128) return;
    float v; int rank;
    if (i < 64) {                          // coarse: linspace(0,2,64)
        v = 2.f * ((float)i / 63.f);
        int c = 0;
        for (int j = 0; j < 64; ++j) {
            float f = 2.f * ((j + 0.5f) / 64.f);
            if (f < v) c++;                // stable: coarse first on ties
        }
        rank = i + c;
    } else {                               // fine: 2*(j+0.5)/64
        int j = i - 64;
        v = 2.f * ((j + 0.5f) / 64.f);
        int c = 0;
        for (int k = 0; k < 64; ++k) {
            float dc = 2.f * ((float)k / 63.f);
            if (dc <= v) c++;
        }
        rank = j + c;
    }
    dall[rank] = v;
}

// ---------------------------------------------------------------------------
// Kernel: convert/repack all weights to bf16 in ws
// ---------------------------------------------------------------------------
__global__ void k_convert(const float* __restrict__ sW,   const float* __restrict__ sWout,
                          const float* __restrict__ rWin, const float* __restrict__ rW,
                          const float* __restrict__ tWin,
                          u16* sW_b, u16* featW_b, float* sdfcol,
                          u16* radInW_b, u16* rW_b, u16* semInW_b) {
    const int tid = blockIdx.x * blockDim.x + threadIdx.x;
    const int nth = gridDim.x * blockDim.x;
    for (int i = tid; i < 7 * 65536; i += nth) sW_b[i] = f2bf(sW[i]);
    for (int i = tid; i < 65536; i += nth) {            // sW_out cols 1..256
        int k = i >> 8, n = i & 255;
        featW_b[i] = f2bf(sWout[k * 257 + 1 + n]);
    }
    for (int i = tid; i < 256; i += nth) sdfcol[i] = sWout[i * 257]; // col 0 (sdf)
    for (int i = tid; i < KPAD * 256; i += nth) {       // rW_in: 265 -> 288 rows
        int r = i >> 8, n = i & 255;
        radInW_b[i] = (r < 265) ? f2bf(rWin[r * 256 + n]) : (u16)0;
    }
    for (int i = tid; i < 2 * 65536; i += nth) rW_b[i] = f2bf(rW[i]);
    for (int i = tid; i < KPAD * 256; i += nth) {
        // tW_in remapped onto the rad_in layout [pts(0-2)|dirs(3-5)|nab(6-8)|feat(9-264)]
        int r = i >> 8, n = i & 255;
        float v;
        if (r < 3)        v = tWin[r * 256 + n];          // pts rows
        else if (r < 9)   v = 0.f;                        // dirs/nablas unused
        else if (r < 265) v = tWin[(r - 6) * 256 + n];    // feat rows
        else              v = 0.f;
        semInW_b[i] = f2bf(v);
    }
}

// ---------------------------------------------------------------------------
// Kernel: ray setup + sample points
// ---------------------------------------------------------------------------
__global__ void k_prep(const float* __restrict__ rays, const float* __restrict__ dall,
                       float* __restrict__ pts, float* __restrict__ dirs) {
    int pt = blockIdx.x * 256 + threadIdx.x;
    if (pt >= NPTS) return;
    int r = pt >> 7, s = pt & 127;
    const float* ry = rays + (size_t)r * 6;
    float o0 = ry[0], o1 = ry[1], o2 = ry[2];
    float d0 = ry[3], d1 = ry[4], d2 = ry[5];
    d0 = (fabsf(d0) < 1e-6f) ? 1e-6f : d0;
    d1 = (fabsf(d1) < 1e-6f) ? 1e-6f : d1;
    d2 = (fabsf(d2) < 1e-6f) ? 1e-6f : d2;
    float inv = 1.f / sqrtf(d0 * d0 + d1 * d1 + d2 * d2);
    d0 *= inv; d1 *= inv; d2 *= inv;
    float t = dall[s];
    pts[(size_t)pt * 3 + 0] = o0 + d0 * t;
    pts[(size_t)pt * 3 + 1] = o1 + d1 * t;
    pts[(size_t)pt * 3 + 2] = o2 + d2 * t;
    if (s == 0) { dirs[r * 3 + 0] = d0; dirs[r * 3 + 1] = d1; dirs[r * 3 + 2] = d2; }
}

// ---------------------------------------------------------------------------
// Main fused MLP kernel: one WG = 32 points (one quarter-ray)
// ---------------------------------------------------------------------------
__global__ void __launch_bounds__(256)
k_main(const float* __restrict__ pts, const float* __restrict__ dirs,
       const float* __restrict__ sWin, const float* __restrict__ sbin,
       const float* __restrict__ sb,   const float* __restrict__ sbout,
       const float* __restrict__ rbin, const float* __restrict__ rb,
       const float* __restrict__ rWout,const float* __restrict__ rbout,
       const float* __restrict__ tbin, const float* __restrict__ tWout,
       const float* __restrict__ tbout,
       const u16* __restrict__ sW_b,   const u16* __restrict__ featW_b,
       const float* __restrict__ sdfcol,
       const u16* __restrict__ radInW_b, const u16* __restrict__ rW_b,
       const u16* __restrict__ semInW_b,
       float* __restrict__ pp) {
    extern __shared__ __align__(128) unsigned char smem[];
    u16*   Wl = (u16*)smem;                       // [288][256] at LDS offset 0
    u16*   b0 = Wl + 288 * 256;                   // [128][288]
    u16*   b1 = b0 + MROWS * KPAD;                // [128][288]
    float* sm = (float*)(b1 + MROWS * KPAD);      // small scratch

    const int tid  = threadIdx.x;
    const int wave = tid >> 5, lane = tid & 31;
    const int P0   = blockIdx.x * PTS_PER_WG;

    // sm[0..95]=pts, sm[96..98]=ray dir, sm[99..130]=sdf, sm[131..226]=nablas
    if (tid < 96) sm[tid] = pts[(size_t)P0 * 3 + tid];
    if (tid < 3)  sm[96 + tid] = dirs[(P0 >> 7) * 3 + tid];
    __syncthreads();

    // ---- SDF layer 1 (3->256) + tangent seeding, VALU (K=3 too small for WMMA)
    for (int idx = tid; idx < PTS_PER_WG * HID; idx += 256) {
        int p = idx >> 8, n = idx & 255;
        float w0 = sWin[n], w1 = sWin[256 + n], w2 = sWin[512 + n];
        float z = sm[p * 3 + 0] * w0 + sm[p * 3 + 1] * w1 + sm[p * 3 + 2] * w2 + sbin[n];
        float h = softplusf(z);
        float m = 1.f - __expf(-h);
        u16* row = b0 + (size_t)(4 * p) * KPAD + n;
        row[0 * KPAD] = f2bf(h);
        row[1 * KPAD] = f2bf(m * w0);   // t = mask * d z/d p_j
        row[2 * KPAD] = f2bf(m * w1);
        row[3 * KPAD] = f2bf(m * w2);
    }
    __syncthreads();

    // ---- 7 hidden SDF layers, value+3 tangents packed in M
    u16 *A = b0, *B = b1;
    for (int L = 0; L < 7; ++L) {
        stage_w(Wl, sW_b + (size_t)L * 65536, 65536, tid, wave);
        __syncthreads();
        wmma_gemm<0, 8>(A, Wl, B, 0, 0, sb + L * HID, 8, wave, lane);
        __syncthreads();
        u16* t = A; A = B; B = t;
    }
    // A (=b1) now holds h8/t8 rows [128][256]

    // compact h8 rows -> B rows 0..31 (feat GEMM input)
    for (int idx = tid; idx < PTS_PER_WG * HID; idx += 256) {
        int p = idx >> 8, k = idx & 255;
        B[(size_t)p * KPAD + k] = A[(size_t)(4 * p) * KPAD + k];
    }
    // sdf + nabla dots against sW_out column 0 (f32)
    if (tid < 128) {
        int p = tid >> 2, w = tid & 3;
        const u16* row = A + (size_t)(4 * p + w) * KPAD;
        float acc = 0.f;
        for (int k = 0; k < HID; ++k) acc += bf2f(row[k]) * sdfcol[k];
        if (w == 0) {
            float s = acc + sbout[0];
            sm[99 + p] = s;
            pp[(size_t)(P0 + p) * 8 + 0] = s;
        } else {
            sm[131 + p * 3 + (w - 1)] = acc;   // nabla component
        }
    }
    __syncthreads();

    // ---- feat = h8 @ sW_out[:,1:] + sb_out[1:]  -> A rows 0..31, cols 9..264
    stage_w(Wl, featW_b, 65536, tid, wave);
    __syncthreads();
    wmma_gemm<2, 8>(B, Wl, A, 9, 0, sbout + 1, 2, wave, lane);
    __syncthreads();

    // fill rad_in extras: [pts | dirs | nablas | feat | 0-pad]
    if (tid < PTS_PER_WG) {
        int p = tid;
        u16* r = A + (size_t)p * KPAD;
        r[0] = f2bf(sm[p * 3 + 0]); r[1] = f2bf(sm[p * 3 + 1]); r[2] = f2bf(sm[p * 3 + 2]);
        r[3] = f2bf(sm[96]); r[4] = f2bf(sm[97]); r[5] = f2bf(sm[98]);
        r[6] = f2bf(sm[131 + p * 3 + 0]);
        r[7] = f2bf(sm[131 + p * 3 + 1]);
        r[8] = f2bf(sm[131 + p * 3 + 2]);
        for (int c2 = 265; c2 < KPAD; ++c2) r[c2] = 0;
    }
    __syncthreads();

    // ---- radiance input layer (K=288): B rows 0..31
    stage_w(Wl, radInW_b, KPAD * 256, tid, wave);
    __syncthreads();
    wmma_gemm<1, 9>(A, Wl, B, 0, 0, rbin, 2, wave, lane);
    __syncthreads();

    // ---- semantic input layer (same A, row-permuted weights): B rows 32..63
    stage_w(Wl, semInW_b, KPAD * 256, tid, wave);
    __syncthreads();
    wmma_gemm<1, 9>(A, Wl, B, 0, 32, tbin, 2, wave, lane);
    __syncthreads();

    // semantic output (256->3), VALU dots
    if (tid < 96) {
        int p = tid / 3, j = tid % 3;
        const u16* row = B + (size_t)(32 + p) * KPAD;
        float acc = tbout[j];
        for (int k = 0; k < HID; ++k) acc += bf2f(row[k]) * tWout[k * 3 + j];
        pp[(size_t)(P0 + p) * 8 + 4 + j] = acc;
    }

    // ---- radiance hidden layers
    stage_w(Wl, rW_b, 65536, tid, wave);
    __syncthreads();
    wmma_gemm<1, 8>(B, Wl, A, 0, 0, rb, 2, wave, lane);
    __syncthreads();
    stage_w(Wl, rW_b + 65536, 65536, tid, wave);
    __syncthreads();
    wmma_gemm<1, 8>(A, Wl, B, 0, 0, rb + HID, 2, wave, lane);
    __syncthreads();

    // radiance output (256->3) + sigmoid, VALU dots
    if (tid < 96) {
        int p = tid / 3, j = tid % 3;
        const u16* row = B + (size_t)p * KPAD;
        float acc = rbout[j];
        for (int k = 0; k < HID; ++k) acc += bf2f(row[k]) * rWout[k * 3 + j];
        pp[(size_t)(P0 + p) * 8 + 1 + j] = 1.f / (1.f + __expf(-acc));
    }
}

// ---------------------------------------------------------------------------
// Kernel: per-ray volume rendering composite
// ---------------------------------------------------------------------------
__global__ void k_comp(const float* __restrict__ pp, const float* __restrict__ dall,
                       const float* __restrict__ lnbeta, float* __restrict__ out) {
    __shared__ float sd[S_SAMP];
    if (threadIdx.x < S_SAMP) sd[threadIdx.x] = dall[threadIdx.x];
    __syncthreads();
    int r = blockIdx.x * 256 + threadIdx.x;
    if (r >= N_RAYS) return;
    float beta = __expf(lnbeta[0] * 10.f);
    float alpha = 1.f / beta;
    float T = 1.f, tsum = 0.f, dsum = 0.f;
    float rgb0 = 0, rgb1 = 0, rgb2 = 0, s0 = 0, s1 = 0, s2 = 0;
    const float* base = pp + (size_t)r * S_SAMP * 8;
    for (int i = 0; i < S_SAMP - 1; ++i) {
        float sdf = base[i * 8];
        float sg = alpha * (sdf > 0.f ? 0.5f * __expf(-sdf / beta)
                                      : 1.f - 0.5f * __expf(sdf / beta));
        float delta = sd[i + 1] - sd[i];
        float x = sg * delta; if (x < 0.f) x = 0.f;
        float p_i = __expf(-x);
        float tau = (1.f - p_i + 1e-10f) * T;
        rgb0 += tau * base[i * 8 + 1];
        rgb1 += tau * base[i * 8 + 2];
        rgb2 += tau * base[i * 8 + 3];
        s0   += tau * base[i * 8 + 4];
        s1   += tau * base[i * 8 + 5];
        s2   += tau * base[i * 8 + 6];
        dsum += tau * sd[i];
        tsum += tau;
        T *= p_i;
    }
    float* o = out + (size_t)r * 8;
    o[0] = rgb0; o[1] = rgb1; o[2] = rgb2;
    o[3] = s0;   o[4] = s1;   o[5] = s2;
    o[6] = dsum / (tsum + 1e-10f);
    o[7] = tsum;
}

// ---------------------------------------------------------------------------
// Host entry
// ---------------------------------------------------------------------------
extern "C" void kernel_launch(void* const* d_in, const int* in_sizes, int n_in,
                              void* d_out, int out_size, void* d_ws, size_t ws_size,
                              hipStream_t stream) {
    (void)in_sizes; (void)n_in; (void)out_size; (void)ws_size;
    const float* rays  = (const float*)d_in[0];
    const float* lnb   = (const float*)d_in[1];
    const float* sWin  = (const float*)d_in[2];
    const float* sbin  = (const float*)d_in[3];
    const float* sW    = (const float*)d_in[4];
    const float* sb    = (const float*)d_in[5];
    const float* sWout = (const float*)d_in[6];
    const float* sbout = (const float*)d_in[7];
    const float* rWin  = (const float*)d_in[8];
    const float* rbin  = (const float*)d_in[9];
    const float* rW    = (const float*)d_in[10];
    const float* rb    = (const float*)d_in[11];
    const float* rWout = (const float*)d_in[12];
    const float* rbout = (const float*)d_in[13];
    const float* tWin  = (const float*)d_in[14];
    const float* tbin  = (const float*)d_in[15];
    const float* tWout = (const float*)d_in[16];
    const float* tbout = (const float*)d_in[17];

    char* ws = (char*)d_ws;
    float* w_dall = (float*)(ws + O_DALL);
    float* w_dir  = (float*)(ws + O_DIR);
    float* w_pts  = (float*)(ws + O_PTS);
    float* w_pp   = (float*)(ws + O_PP);
    u16*   sW_b     = (u16*)(ws + O_SWB);
    u16*   featW_b  = (u16*)(ws + O_FEATW);
    float* sdfcol   = (float*)(ws + O_SDFCOL);
    u16*   radInW_b = (u16*)(ws + O_RADINW);
    u16*   rW_b     = (u16*)(ws + O_RWB);
    u16*   semInW_b = (u16*)(ws + O_SEMINW);

    (void)hipFuncSetAttribute(reinterpret_cast<const void*>(k_main),
                              hipFuncAttributeMaxDynamicSharedMemorySize, SMEM_BYTES);

    k_dall<<<1, 128, 0, stream>>>(w_dall);
    k_convert<<<256, 256, 0, stream>>>(sW, sWout, rWin, rW, tWin,
                                       sW_b, featW_b, sdfcol, radInW_b, rW_b, semInW_b);
    k_prep<<<NPTS / 256, 256, 0, stream>>>(rays, w_dall, w_pts, w_dir);
    k_main<<<NPTS / PTS_PER_WG, 256, SMEM_BYTES, stream>>>(
        w_pts, w_dir, sWin, sbin, sb, sbout, rbin, rb, rWout, rbout,
        tbin, tWout, tbout, sW_b, featW_b, sdfcol, radInW_b, rW_b, semInW_b, w_pp);
    k_comp<<<N_RAYS / 256, 256, 0, stream>>>(w_pp, w_dall, lnb, (float*)d_out);
}
